// Graph_Net_art_4174708212320
// MI455X (gfx1250) — compile-verified, matrix operand order checked by reading, not run on hardware
//
#include <hip/hip_runtime.h>

// ---------------------------------------------------------------------------
// MI455X / gfx1250 GNN (graph coloring) forward pass.
// All GEMMs -> v_wmma_f32_16x16x32_bf16 (wave32). Adj converted to bf16 once
// (128 MB, L2-resident for all 32 iterations). B panel (VhT) staged per-block
// into LDS with double-buffered global_load_async_to_lds_b128 (ASYNCcnt),
// killing the 2x redundant L2 traffic. LDS row strides padded so b128 frag
// loads start at bank 4*lane (conflict-free 2-cycle service).
// ---------------------------------------------------------------------------

typedef __bf16 bf16_t;
typedef bf16_t v16bf __attribute__((ext_vector_type(16)));
typedef bf16_t v8bf  __attribute__((ext_vector_type(8)));
typedef bf16_t v4bf  __attribute__((ext_vector_type(4)));
typedef float  v8f   __attribute__((ext_vector_type(8)));
typedef float  v4f   __attribute__((ext_vector_type(4)));

#define NV   8192   // total vertices
#define NCOL 256    // total colors
#define DD   32     // MLP_DIM
#define KC   512    // B-panel k-chunk staged in LDS
#define BPLD 520    // padded B-panel row stride (1040B: start bank = 4*lane)
#define STLD 72     // padded 16x64 staging tile stride
#define MLLD 40     // padded 16x32 staging tile stride

__device__ __forceinline__ v8f zero8() {
  v8f z;
#pragma unroll
  for (int r = 0; r < 8; ++r) z[r] = 0.0f;
  return z;
}

__device__ __forceinline__ v8f wmma_bf16(v16bf a, v16bf b, v8f c) {
  // emits v_wmma_f32_16x16x32_bf16
  return __builtin_amdgcn_wmma_f32_16x16x32_bf16(false, a, false, b, (short)0, c,
                                                 false, false);
}

// Async copy of 16B global -> LDS (per lane). Tracked by ASYNCcnt.
__device__ __forceinline__ void async_b128(bf16_t* lds_dst, const bf16_t* gsrc) {
  unsigned l = (unsigned)(uintptr_t)lds_dst;  // low 32 bits = LDS offset
  asm volatile("global_load_async_to_lds_b128 %0, %1, off"
               :: "v"(l), "v"(gsrc) : "memory");
}

// A or B fragment for 16x16x32 bf16 WMMA from a K-contiguous row-major matrix.
// Lane L: row (A: M, B: N) = L&15, h = L>>4; elems 0..7 = K[8h..8h+7],
// elems 8..15 = K[16+8h..16+8h+7]  -> two b128 loads per lane.
__device__ __forceinline__ v16bf load_frag_k32(const bf16_t* base, int ld,
                                               int row0, int k0, int lane) {
  int m = lane & 15, h = lane >> 4;
  const bf16_t* p = base + (size_t)(row0 + m) * (size_t)ld + k0;
  union { v8bf h8[2]; v16bf w; } u;
  u.h8[0] = *(const v8bf*)(p + 8 * h);
  u.h8[1] = *(const v8bf*)(p + 16 + 8 * h);
  return u.w;
}

// C/D fragment (v8f): lane L holds col n = L&15, rows m0 + 8*(L>>4) + r.
__device__ __forceinline__ v8f load_tile_f32(const float* src, int ld, int m0,
                                             int n0, int lane) {
  int n = lane & 15, h = lane >> 4;
  v8f c;
#pragma unroll
  for (int r = 0; r < 8; ++r)
    c[r] = src[(size_t)(m0 + 8 * h + r) * ld + n0 + n];
  return c;
}

__device__ __forceinline__ void store_tile_f32(float* dst, int ld, int m0,
                                               int n0, v8f v, int lane) {
  int n = lane & 15, h = lane >> 4;
#pragma unroll
  for (int r = 0; r < 8; ++r)
    dst[(size_t)(m0 + 8 * h + r) * ld + n0 + n] = v[r];
}

__device__ __forceinline__ void store_tile_bf16(bf16_t* dst, int ld, int m0,
                                                int n0, v8f v, int lane) {
  int n = lane & 15, h = lane >> 4;
#pragma unroll
  for (int r = 0; r < 8; ++r)
    dst[(size_t)(m0 + 8 * h + r) * ld + n0 + n] = (bf16_t)v[r];
}

// Transposed bf16 store: per lane 8 consecutive rows -> one 16B store.
__device__ __forceinline__ void store_tile_T_bf16(bf16_t* dstT, int ldT, int m0,
                                                  int n0, v8f v, int lane) {
  int n = lane & 15, h = lane >> 4;
  v8bf o;
#pragma unroll
  for (int r = 0; r < 8; ++r) o[r] = (bf16_t)v[r];
  *(v8bf*)(dstT + (size_t)(n0 + n) * (size_t)ldT + m0 + 8 * h) = o;
}

// C-layout -> row-major bf16 LDS staging (for later A-fragment reload).
__device__ __forceinline__ void store_tile_lds(bf16_t* t, int ldt, v8f v,
                                               int n0, int lane) {
  int n = lane & 15, h = lane >> 4;
#pragma unroll
  for (int r = 0; r < 8; ++r) t[(8 * h + r) * ldt + n0 + n] = (bf16_t)v[r];
}

__device__ __forceinline__ float sigm(float x) {
  return 1.0f / (1.0f + __expf(-x));
}

// LSTM elementwise tail. g[0..7] gate N-tiles: i=0,1 f=2,3 g~=4,5 o=6,7.
__device__ __forceinline__ void lstm_tail(v8f* g, const float* bih,
                                          const float* bhh, float* Cst,
                                          bf16_t* Hrow, bf16_t* HT, int ldT,
                                          int m0, int lane) {
  int nn = lane & 15;
#pragma unroll
  for (int p = 0; p < 2; ++p) {
    int col = p * 16 + nn;
    float bi = bih[col] + bhh[col];
    float bf_ = bih[32 + col] + bhh[32 + col];
    float bg = bih[64 + col] + bhh[64 + col];
    float bo = bih[96 + col] + bhh[96 + col];
    v8f c_old = load_tile_f32(Cst, DD, m0, p * 16, lane);
    v8f cn, hn;
#pragma unroll
    for (int r = 0; r < 8; ++r) {
      float iv = sigm(g[p][r] + bi);
      float fv = sigm(g[2 + p][r] + bf_);
      float gv = tanhf(g[4 + p][r] + bg);
      float ov = sigm(g[6 + p][r] + bo);
      float c2 = fv * c_old[r] + iv * gv;
      cn[r] = c2;
      hn[r] = ov * tanhf(c2);
    }
    store_tile_f32(Cst, DD, m0, p * 16, cn, lane);
    store_tile_bf16(Hrow, DD, m0, p * 16, hn, lane);
    if (HT) store_tile_T_bf16(HT, ldT, m0, p * 16, hn, lane);
  }
}

// ------------------------------ prep kernels -------------------------------

__global__ void k_cvt4(const float* __restrict__ src, bf16_t* __restrict__ dst,
                       long n4) {
  long i = (long)blockIdx.x * blockDim.x + threadIdx.x;
  if (i < n4) {
    v4f v = ((const v4f*)src)[i];
    v4bf o;
#pragma unroll
    for (int j = 0; j < 4; ++j) o[j] = (bf16_t)v[j];
    ((v4bf*)dst)[i] = o;
  }
}

__global__ void k_mvct(const float* __restrict__ Mvc, bf16_t* __restrict__ MvcT) {
  long i = (long)blockIdx.x * blockDim.x + threadIdx.x;
  if (i < (long)NV * NCOL) {
    int r = (int)(i / NCOL), c = (int)(i % NCOL);
    MvcT[(size_t)c * NV + r] = (bf16_t)Mvc[i];
  }
}

__global__ void k_init(const float* __restrict__ WvI, const float* __restrict__ bvI,
                       const float* __restrict__ WcI, const float* __restrict__ bcI,
                       bf16_t* __restrict__ VhB, bf16_t* __restrict__ VhT,
                       float* __restrict__ Vc, bf16_t* __restrict__ ChB,
                       float* __restrict__ Cc) {
  int i = blockIdx.x * blockDim.x + threadIdx.x;
  if (i < NV) {
#pragma unroll
    for (int j = 0; j < DD; ++j) {
      float v = WvI[j] + bvI[j];  // linear([1.0], (W(32,1), b))
      bf16_t b = (bf16_t)v;
      VhB[(size_t)i * DD + j] = b;
      VhT[(size_t)j * NV + i] = b;
      Vc[(size_t)i * DD + j] = 0.0f;
    }
  }
  if (i < NCOL) {
#pragma unroll
    for (int j = 0; j < DD; ++j) {
      float v = WcI[j] + bcI[j];
      ChB[(size_t)i * DD + j] = (bf16_t)v;
      Cc[(size_t)i * DD + j] = 0.0f;
    }
  }
}

// ------------------------- K1: both 3-layer MLPs ---------------------------
// 132 blocks x 128 threads; wave-tile t<512 -> vertex MLP, else color MLP.
__global__ void k_mlp(const bf16_t* __restrict__ VhB, const bf16_t* __restrict__ ChB,
                      const bf16_t* __restrict__ WvB, const bf16_t* __restrict__ WcB,
                      const float* __restrict__ bv0, const float* __restrict__ bv1,
                      const float* __restrict__ bv2, const float* __restrict__ bc0,
                      const float* __restrict__ bc1, const float* __restrict__ bc2,
                      bf16_t* __restrict__ VmlpT, bf16_t* __restrict__ CmlpT) {
  __shared__ __align__(16) bf16_t lds[4][16 * MLLD];
  int wave = threadIdx.x >> 5, lane = threadIdx.x & 31;
  int t = blockIdx.x * 4 + wave;
  bool isV = (t < NV / 16);
  const bf16_t* src = isV ? VhB : ChB;
  const bf16_t* W = isV ? WvB : WcB;
  const float* bl[3] = {isV ? bv0 : bc0, isV ? bv1 : bc1, isV ? bv2 : bc2};
  bf16_t* dstT = isV ? VmlpT : CmlpT;
  int ldT = isV ? NV : NCOL;
  int m0 = (isV ? t : (t - NV / 16)) * 16;
  bf16_t* tl = &lds[wave][0];
  int nn = lane & 15;

  v16bf A = load_frag_k32(src, DD, m0, 0, lane);
#pragma unroll
  for (int l = 0; l < 3; ++l) {
    v16bf B0 = load_frag_k32(W + l * 1024, DD, 0, 0, lane);
    v16bf B1 = load_frag_k32(W + l * 1024, DD, 16, 0, lane);
    v8f c0 = wmma_bf16(A, B0, zero8());
    v8f c1 = wmma_bf16(A, B1, zero8());
    float bb0 = bl[l][nn], bb1 = bl[l][16 + nn];
#pragma unroll
    for (int r = 0; r < 8; ++r) {
      c0[r] += bb0;
      c1[r] += bb1;
      if (l < 2) { c0[r] = fmaxf(c0[r], 0.0f); c1[r] = fmaxf(c1[r], 0.0f); }
    }
    if (l < 2) {
      store_tile_lds(tl, MLLD, c0, 0, lane);
      store_tile_lds(tl, MLLD, c1, 16, lane);
      asm volatile("s_wait_dscnt 0x0" ::: "memory");
      A = load_frag_k32(tl, MLLD, 0, 0, lane);
    } else {
      store_tile_T_bf16(dstT, ldT, m0, 0, c0, lane);
      store_tile_T_bf16(dstT, ldT, m0, 16, c1, lane);
    }
  }
}

// --------------- K2: Adj@Vh + Mvc@Cmlp + vertex LSTM (fused) ---------------
// 128 blocks x 128 threads; one wave per 16-row tile. VhT B-panel staged in
// LDS via double-buffered async copies.
__global__ void k_vertex(const bf16_t* __restrict__ AdjB, const bf16_t* __restrict__ MvcB,
                         const bf16_t* __restrict__ VhT_c, const bf16_t* __restrict__ VhB_c,
                         const bf16_t* __restrict__ CmlpT,
                         const bf16_t* __restrict__ WihV, const bf16_t* __restrict__ WhhV,
                         const float* __restrict__ bih, const float* __restrict__ bhh,
                         float* __restrict__ Vc, bf16_t* __restrict__ VhB_n,
                         bf16_t* __restrict__ VhT_n) {
  __shared__ __align__(16) bf16_t Bp[2][DD * BPLD];   // 2 x 32.5KB B panels
  __shared__ __align__(16) bf16_t stg[4][16 * STLD];  // gate staging tiles
  int wave = threadIdx.x >> 5, lane = threadIdx.x & 31;
  int m0 = (blockIdx.x * 4 + wave) * 16;
  int mm = lane & 15;
  const bf16_t* arow = AdjB + (size_t)(m0 + mm) * NV;

  // cooperative async copy: thread t -> row t/4, quarter t%4 of 32 x KC panel
  int rr = threadIdx.x >> 2, qq = threadIdx.x & 3;
  auto panel_issue = [&](bf16_t* dbuf, int k0) {
    const bf16_t* g = VhT_c + (size_t)rr * NV + k0 + qq * (KC / 4);
    bf16_t* d = dbuf + rr * BPLD + qq * (KC / 4);
#pragma unroll
    for (int u = 0; u < (KC / 4) / 8; ++u)  // 16 x b128 per lane
      async_b128(d + u * 8, g + u * 8);
  };

  // AdjVh: K = 8192 streamed from L2-resident bf16 Adj; B from LDS panel.
  v8f a0 = zero8(), a1 = zero8();
  const int NCH = NV / KC;  // 16 chunks
  panel_issue(&Bp[0][0], 0);
  for (int c = 0; c < NCH; ++c) {
    if (c + 1 < NCH) {
      panel_issue(&Bp[(c + 1) & 1][0], (c + 1) * KC);
      asm volatile("s_wait_asynccnt 0x10" ::: "memory");  // chunk c done
    } else {
      asm volatile("s_wait_asynccnt 0x0" ::: "memory");
    }
    __syncthreads();
    const bf16_t* B = &Bp[c & 1][0];
#pragma unroll 4
    for (int k = 0; k < KC; k += 32) {
      __builtin_prefetch(arow + c * KC + k + 1024, 0, 1);  // global_prefetch
      v16bf A = load_frag_k32(AdjB, NV, m0, c * KC + k, lane);
      v16bf B0 = load_frag_k32(B, BPLD, 0, k, lane);   // ds_load_b128 x2
      v16bf B1 = load_frag_k32(B, BPLD, 16, k, lane);
      a0 = wmma_bf16(A, B0, a0);
      a1 = wmma_bf16(A, B1, a1);
    }
    __syncthreads();  // all waves done reading before panel reuse
  }

  // Vmsg = Mvc @ Cmlp : K = 256 (small, straight from global/L2).
  v8f v0 = zero8(), v1 = zero8();
#pragma unroll
  for (int k = 0; k < NCOL; k += 32) {
    v16bf A = load_frag_k32(MvcB, NCOL, m0, k, lane);
    v16bf B0 = load_frag_k32(CmlpT, NCOL, 0, k, lane);
    v16bf B1 = load_frag_k32(CmlpT, NCOL, 16, k, lane);
    v0 = wmma_bf16(A, B0, v0);
    v1 = wmma_bf16(A, B1, v1);
  }

  // rnn_inp tile (16x64) -> per-wave LDS (C-layout -> A-layout relayout).
  bf16_t* tl = &stg[wave][0];
  store_tile_lds(tl, STLD, a0, 0, lane);
  store_tile_lds(tl, STLD, a1, 16, lane);
  store_tile_lds(tl, STLD, v0, 32, lane);
  store_tile_lds(tl, STLD, v1, 48, lane);
  asm volatile("s_wait_dscnt 0x0" ::: "memory");

  // gates: rnn_inp @ WihV.T (K=64) + Vh @ WhhV.T (K=32), N = 128 (8 tiles).
  v8f g[8];
#pragma unroll
  for (int j = 0; j < 8; ++j) g[j] = zero8();
#pragma unroll
  for (int kk = 0; kk < 64; kk += 32) {
    v16bf A = load_frag_k32(tl, STLD, 0, kk, lane);
#pragma unroll
    for (int j = 0; j < 8; ++j) {
      v16bf B = load_frag_k32(WihV, 64, j * 16, kk, lane);
      g[j] = wmma_bf16(A, B, g[j]);
    }
  }
  {
    v16bf Ah = load_frag_k32(VhB_c, DD, m0, 0, lane);
#pragma unroll
    for (int j = 0; j < 8; ++j) {
      v16bf B = load_frag_k32(WhhV, DD, j * 16, 0, lane);
      g[j] = wmma_bf16(Ah, B, g[j]);
    }
  }
  lstm_tail(g, bih, bhh, Vc, VhB_n, VhT_n, NV, m0, lane);
}

// ----------------- K3: MvcT@Vmlp + color LSTM (fused) ----------------------
// 4 blocks x 128 threads.
__global__ void k_color(const bf16_t* __restrict__ MvcT, const bf16_t* __restrict__ VmlpT,
                        const bf16_t* __restrict__ ChB_c,
                        const bf16_t* __restrict__ WihC, const bf16_t* __restrict__ WhhC,
                        const float* __restrict__ bih, const float* __restrict__ bhh,
                        float* __restrict__ Cc, bf16_t* __restrict__ ChB_n) {
  __shared__ __align__(16) bf16_t lds[4][16 * MLLD];
  int wave = threadIdx.x >> 5, lane = threadIdx.x & 31;
  int m0 = (blockIdx.x * 4 + wave) * 16;
  v8f a0 = zero8(), a1 = zero8();
  for (int k = 0; k < NV; k += 32) {
    v16bf A = load_frag_k32(MvcT, NV, m0, k, lane);
    v16bf B0 = load_frag_k32(VmlpT, NV, 0, k, lane);
    v16bf B1 = load_frag_k32(VmlpT, NV, 16, k, lane);
    a0 = wmma_bf16(A, B0, a0);
    a1 = wmma_bf16(A, B1, a1);
  }
  bf16_t* tl = &lds[wave][0];
  store_tile_lds(tl, MLLD, a0, 0, lane);
  store_tile_lds(tl, MLLD, a1, 16, lane);
  asm volatile("s_wait_dscnt 0x0" ::: "memory");
  v16bf Ax = load_frag_k32(tl, MLLD, 0, 0, lane);
  v16bf Ah = load_frag_k32(ChB_c, DD, m0, 0, lane);
  v8f g[8];
#pragma unroll
  for (int j = 0; j < 8; ++j) {
    v16bf Bx = load_frag_k32(WihC, DD, j * 16, 0, lane);
    v16bf Bh = load_frag_k32(WhhC, DD, j * 16, 0, lane);
    g[j] = wmma_bf16(Ax, Bx, zero8());
    g[j] = wmma_bf16(Ah, Bh, g[j]);
  }
  lstm_tail(g, bih, bhh, Cc, ChB_n, (bf16_t*)nullptr, 0, m0, lane);
}

// ------------------- K4: vote MLP + per-graph mean + sigmoid ---------------
__global__ void k_vote(const bf16_t* __restrict__ VhB, const float* __restrict__ W0,
                       const float* __restrict__ b0, const float* __restrict__ W1,
                       const float* __restrict__ b1, const float* __restrict__ W2,
                       const float* __restrict__ b2, float* __restrict__ out) {
  __shared__ float red[256];
  int gi = blockIdx.x, t = threadIdx.x;
  int row = gi * 256 + t;
  float x[DD], y[DD], z[DD];
#pragma unroll
  for (int j = 0; j < DD; ++j) x[j] = (float)VhB[(size_t)row * DD + j];
  for (int o = 0; o < DD; ++o) {
    float s = b0[o];
#pragma unroll
    for (int j = 0; j < DD; ++j) s += W0[o * DD + j] * x[j];
    y[o] = s;
  }
  for (int o = 0; o < DD; ++o) {
    float s = b1[o];
#pragma unroll
    for (int j = 0; j < DD; ++j) s += W1[o * DD + j] * y[j];
    z[o] = s;
  }
  float lg = b2[0];
#pragma unroll
  for (int j = 0; j < DD; ++j) lg += W2[j] * z[j];
  red[t] = lg;
  __syncthreads();
  for (int s = 128; s > 0; s >>= 1) {
    if (t < s) red[t] += red[t + s];
    __syncthreads();
  }
  if (t == 0) out[gi] = 1.0f / (1.0f + __expf(-red[0] / 256.0f));
}

// ---------------------------------------------------------------------------
extern "C" void kernel_launch(void* const* d_in, const int* in_sizes, int n_in,
                              void* d_out, int out_size, void* d_ws, size_t ws_size,
                              hipStream_t stream) {
  (void)in_sizes; (void)n_in; (void)out_size;
  // Input order: setup_inputs() insertion order; `params` flattened jax-tree
  // style (dict keys sorted, ASCII: uppercase < lowercase):
  //  0 Adj  1 Mvc
  //  C_init: 2 W 3 b | C_mlp: 4..9 (W,b)x3 | C_update: 10 Whh 11 Wih 12 bhh 13 bih
  //  V_init: 14 W 15 b | V_mlp: 16..21      | V_update: 22 Whh 23 Wih 24 bhh 25 bih
  //  vote: 26 W0 27 b0 28 W1 29 b1 30 W2 31 b2 | 32 nvert 33 ncol
  const float* Adj = (const float*)d_in[0];
  const float* Mvc = (const float*)d_in[1];
  const float* CinW = (const float*)d_in[2];
  const float* Cinb = (const float*)d_in[3];
  const float* CmlpW[3] = {(const float*)d_in[4], (const float*)d_in[6], (const float*)d_in[8]};
  const float* Cmlpb[3] = {(const float*)d_in[5], (const float*)d_in[7], (const float*)d_in[9]};
  const float* CWhh = (const float*)d_in[10];
  const float* CWih = (const float*)d_in[11];
  const float* Cbhh = (const float*)d_in[12];
  const float* Cbih = (const float*)d_in[13];
  const float* VinW = (const float*)d_in[14];
  const float* Vinb = (const float*)d_in[15];
  const float* VmlpW[3] = {(const float*)d_in[16], (const float*)d_in[18], (const float*)d_in[20]};
  const float* Vmlpb[3] = {(const float*)d_in[17], (const float*)d_in[19], (const float*)d_in[21]};
  const float* VWhh = (const float*)d_in[22];
  const float* VWih = (const float*)d_in[23];
  const float* Vbhh = (const float*)d_in[24];
  const float* Vbih = (const float*)d_in[25];
  const float* voteW[3] = {(const float*)d_in[26], (const float*)d_in[28], (const float*)d_in[30]};
  const float* voteb[3] = {(const float*)d_in[27], (const float*)d_in[29], (const float*)d_in[31]};

  // workspace layout
  char* w = (char*)d_ws;
  size_t off = 0;
  auto take = [&](size_t bytes) { size_t o = off; off += (bytes + 255) & ~(size_t)255; return o; };
  bf16_t* AdjB   = (bf16_t*)(w + take((size_t)NV * NV * 2));
  bf16_t* MvcB   = (bf16_t*)(w + take((size_t)NV * NCOL * 2));
  bf16_t* MvcT   = (bf16_t*)(w + take((size_t)NV * NCOL * 2));
  bf16_t* VhB[2] = {(bf16_t*)(w + take((size_t)NV * DD * 2)),
                    (bf16_t*)(w + take((size_t)NV * DD * 2))};
  bf16_t* VhT[2] = {(bf16_t*)(w + take((size_t)NV * DD * 2)),
                    (bf16_t*)(w + take((size_t)NV * DD * 2))};
  bf16_t* ChB[2] = {(bf16_t*)(w + take((size_t)NCOL * DD * 2)),
                    (bf16_t*)(w + take((size_t)NCOL * DD * 2))};
  bf16_t* VmlpT  = (bf16_t*)(w + take((size_t)NV * DD * 2));
  bf16_t* CmlpT  = (bf16_t*)(w + take((size_t)NCOL * DD * 2));
  float*  Vc     = (float*)(w + take((size_t)NV * DD * 4));
  float*  Cc     = (float*)(w + take((size_t)NCOL * DD * 4));
  bf16_t* WvB    = (bf16_t*)(w + take(3 * 1024 * 2));
  bf16_t* WcB    = (bf16_t*)(w + take(3 * 1024 * 2));
  bf16_t* WihVB  = (bf16_t*)(w + take(128 * 64 * 2));
  bf16_t* WhhVB  = (bf16_t*)(w + take(128 * 32 * 2));
  bf16_t* WihCB  = (bf16_t*)(w + take(128 * 32 * 2));
  bf16_t* WhhCB  = (bf16_t*)(w + take(128 * 32 * 2));
  if (ws_size < off) return;  // workspace too small: bail deterministically

  auto cvt = [&](const float* s, bf16_t* d, long n) {
    long n4 = n / 4;
    k_cvt4<<<(int)((n4 + 255) / 256), 256, 0, stream>>>(s, d, n4);
  };
  // one-time conversions (Adj: 256MB f32 -> 128MB bf16, fits 192MB L2)
  cvt(Adj, AdjB, (long)NV * NV);
  cvt(Mvc, MvcB, (long)NV * NCOL);
  k_mvct<<<(int)(((long)NV * NCOL + 255) / 256), 256, 0, stream>>>(Mvc, MvcT);
  for (int l = 0; l < 3; ++l) {
    cvt(VmlpW[l], WvB + l * 1024, 1024);
    cvt(CmlpW[l], WcB + l * 1024, 1024);
  }
  cvt(VWih, WihVB, 128 * 64);
  cvt(VWhh, WhhVB, 128 * 32);
  cvt(CWih, WihCB, 128 * 32);
  cvt(CWhh, WhhCB, 128 * 32);
  k_init<<<(NV + 255) / 256, 256, 0, stream>>>(VinW, Vinb, CinW, Cinb, VhB[0],
                                               VhT[0], Vc, ChB[0], Cc);

  for (int it = 0; it < 32; ++it) {
    int cur = it & 1, nxt = cur ^ 1;
    k_mlp<<<132, 128, 0, stream>>>(VhB[cur], ChB[cur], WvB, WcB, Vmlpb[0],
                                   Vmlpb[1], Vmlpb[2], Cmlpb[0], Cmlpb[1],
                                   Cmlpb[2], VmlpT, CmlpT);
    k_vertex<<<128, 128, 0, stream>>>(AdjB, MvcB, VhT[cur], VhB[cur], CmlpT,
                                      WihVB, WhhVB, Vbih, Vbhh, Vc, VhB[nxt],
                                      VhT[nxt]);
    k_color<<<4, 128, 0, stream>>>(MvcT, VmlpT, ChB[cur], WihCB, WhhCB, Cbih,
                                   Cbhh, Cc, ChB[nxt]);
  }
  // after 32 iters the final Vh is in buffer 0
  k_vote<<<32, 256, 0, stream>>>(VhB[0], voteW[0], voteb[0], voteW[1], voteb[1],
                                 voteW[2], voteb[2], (float*)d_out);
}
